// stagenet_35210141892752
// MI455X (gfx1250) — matrix operand order repaired and checked by reading, not compiled, not run on hardware
//
#include <hip/hip_runtime.h>
#include <math.h>

// ---------------------------------------------------------------------------
// MVS group-correlation + depth regression for MI455X (gfx1250, wave32).
//   - one wave == one pixel, lane == depth hypothesis (D=32 == wave32)
//   - channel contraction (cor + cor.sum) done with v_wmma_f32_16x16x32_f16
//   - bilinear gathers stay on VMEM (all features L2-resident: 21MB << 192MB)
//   - wave reductions via DPP8 / DPP16 row_ror:8 / ds_swizzle SWAPX16
// ---------------------------------------------------------------------------

typedef _Float16 v16h __attribute__((ext_vector_type(16)));
typedef _Float16 v8h  __attribute__((ext_vector_type(8)));
typedef float    v8f  __attribute__((ext_vector_type(8)));
typedef float    v4f  __attribute__((ext_vector_type(4)));

#define B_  2
#define V_  4
#define C_  32
#define D_  32
#define H_  128
#define W_  160
#define G_  8
#define HW_ (H_*W_)

// ---- wave32 butterfly reduction helpers (VALU crossbar + 1 DS swizzle) ----
static constexpr unsigned dpp8_xor_sel(int k) {
  unsigned s = 0;
  for (int i = 0; i < 8; ++i) s |= (unsigned)((i ^ k) << (3 * i));
  return s;
}
static constexpr unsigned DPP8_X1 = dpp8_xor_sel(1);
static constexpr unsigned DPP8_X2 = dpp8_xor_sel(2);
static constexpr unsigned DPP8_X4 = dpp8_xor_sel(4);
#define DPP_ROW_ROR8   0x128   // DPP16 row_ror:8  (xor-8 for commutative red.)
#define SWZ_SWAPX16    0x401F  // ds_swizzle group-of-32: xor_mask=0x10, and=0x1f

__device__ __forceinline__ float wred_max_f32(float x) {
  float t;
  t = __int_as_float(__builtin_amdgcn_mov_dpp8(__float_as_int(x), DPP8_X1));
  x = fmaxf(x, t);
  t = __int_as_float(__builtin_amdgcn_mov_dpp8(__float_as_int(x), DPP8_X2));
  x = fmaxf(x, t);
  t = __int_as_float(__builtin_amdgcn_mov_dpp8(__float_as_int(x), DPP8_X4));
  x = fmaxf(x, t);
  t = __int_as_float(__builtin_amdgcn_update_dpp(__float_as_int(x), __float_as_int(x),
                                                 DPP_ROW_ROR8, 0xf, 0xf, false));
  x = fmaxf(x, t);
  t = __int_as_float(__builtin_amdgcn_ds_swizzle(__float_as_int(x), SWZ_SWAPX16));
  x = fmaxf(x, t);
  return x;
}

__device__ __forceinline__ float wred_sum_f32(float x) {
  float t;
  t = __int_as_float(__builtin_amdgcn_mov_dpp8(__float_as_int(x), DPP8_X1));
  x += t;
  t = __int_as_float(__builtin_amdgcn_mov_dpp8(__float_as_int(x), DPP8_X2));
  x += t;
  t = __int_as_float(__builtin_amdgcn_mov_dpp8(__float_as_int(x), DPP8_X4));
  x += t;
  t = __int_as_float(__builtin_amdgcn_update_dpp(__float_as_int(x), __float_as_int(x),
                                                 DPP_ROW_ROR8, 0xf, 0xf, false));
  x += t;
  t = __int_as_float(__builtin_amdgcn_ds_swizzle(__float_as_int(x), SWZ_SWAPX16));
  x += t;
  return x;
}

// ---- tiny projection prep: rot/trans of src_proj @ inv(ref_proj) ----------
__device__ inline void make_proj3(const float* P, int b, int vi,
                                  float M[9], float t[3]) {
  const float* ext = P + (((size_t)(b*V_ + vi))*2 + 0)*16;
  const float* K   = P + (((size_t)(b*V_ + vi))*2 + 1)*16;
  // top 3x4 of make_proj: K[:3,:3] @ ext[:3,:4]
  for (int r = 0; r < 3; ++r) {
    for (int c = 0; c < 3; ++c) {
      float s = 0.f;
      for (int j = 0; j < 3; ++j) s += K[r*4+j] * ext[j*4+c];
      M[r*3+c] = s;
    }
    float s = 0.f;
    for (int j = 0; j < 3; ++j) s += K[r*4+j] * ext[j*4+3];
    t[r] = s;
  }
}

__global__ void proj_prep(const float* __restrict__ P, float* __restrict__ ws) {
  int t = threadIdx.x;
  if (t >= B_*(V_-1)) return;
  int b = t / (V_-1);
  int v = t % (V_-1) + 1;

  float Mr[9], tr[3], Ms[9], ts[3];
  make_proj3(P, b, 0, Mr, tr);
  make_proj3(P, b, v, Ms, ts);

  // inv(Mr) via adjugate (both projections have last row [0,0,0,1], so the
  // 4x4 inverse in the reference reduces to this block form).
  float c00 = Mr[4]*Mr[8] - Mr[5]*Mr[7];
  float c01 = Mr[5]*Mr[6] - Mr[3]*Mr[8];
  float c02 = Mr[3]*Mr[7] - Mr[4]*Mr[6];
  float det = Mr[0]*c00 + Mr[1]*c01 + Mr[2]*c02;
  float id  = 1.f / det;
  float Ri[9];
  Ri[0]=c00*id;                     Ri[1]=(Mr[2]*Mr[7]-Mr[1]*Mr[8])*id; Ri[2]=(Mr[1]*Mr[5]-Mr[2]*Mr[4])*id;
  Ri[3]=c01*id;                     Ri[4]=(Mr[0]*Mr[8]-Mr[2]*Mr[6])*id; Ri[5]=(Mr[2]*Mr[3]-Mr[0]*Mr[5])*id;
  Ri[6]=c02*id;                     Ri[7]=(Mr[1]*Mr[6]-Mr[0]*Mr[7])*id; Ri[8]=(Mr[0]*Mr[4]-Mr[1]*Mr[3])*id;

  float R[9];
  for (int r = 0; r < 3; ++r)
    for (int c = 0; c < 3; ++c)
      R[r*3+c] = Ms[r*3+0]*Ri[0*3+c] + Ms[r*3+1]*Ri[1*3+c] + Ms[r*3+2]*Ri[2*3+c];
  float T[3];
  for (int r = 0; r < 3; ++r)
    T[r] = ts[r] - (R[r*3+0]*tr[0] + R[r*3+1]*tr[1] + R[r*3+2]*tr[2]);

  float* o = ws + t*12;
  for (int i = 0; i < 9; ++i) o[i] = R[i];
  o[9] = T[0]; o[10] = T[1]; o[11] = T[2];
}

// ---- main kernel ----------------------------------------------------------
__global__ void __launch_bounds__(256) mvs_main(
    const float* __restrict__ ref,   // (B,C,H,W)
    const float* __restrict__ src,   // (V-1,B,C,H,W)
    const float* __restrict__ dh,    // (B,D,H,W)
    const float* __restrict__ regw,  // (8,)
    const float* __restrict__ ws,    // 6 x (rot9 + trans3)
    float*       __restrict__ out)   // depth (B,H,W) ++ attn (B,D,H,W)
{
  // per-wave private LDS slices (no cross-wave sharing -> no barriers;
  // LDS ops are in-order within a wave, explicit s_wait_dscnt for safety)
  __shared__ __align__(32) _Float16 Ast[8][D_*C_];  // 8 waves * 2KB
  __shared__ __align__(16) float    Cst[8][D_*16];  // 8 waves * 2KB

  const int tid  = threadIdx.x;
  const int lane = tid & 31;
  const int wv   = tid >> 5;
  const int pid  = blockIdx.x * 8 + wv;       // grid covers exactly B*H*W
  const int b    = pid / HW_;
  const int pix  = pid % HW_;
  const int hy   = pix / W_;
  const int wx   = pix % W_;

  // lane == depth index
  const float dep = dh[((size_t)(b*D_ + lane))*HW_ + pix];

  // ---- build B matrix once per pixel: masked ref/4 ------------------------
  // B 32x16, contiguous k order used here (same as A): lane L element e -> B[k = e + (L<16?0:16)][n = L&15]
  //   col g (g<8): channels of group g (group mean => *0.25)
  //   col 8      : all channels (cor.sum(axis=1)) ; cols 9..15 zero
  const int nCol  = lane & 15;
  const int kBase = (lane < 16) ? 0 : 16;
  v16h bmat;
  #pragma unroll
  for (int e = 0; e < 16; ++e) {
    int   k  = kBase + e;
    float rv = ref[((size_t)(b*C_ + k))*HW_ + pix];
    float v  = 0.f;
    if (nCol < 8)       { if ((k >> 2) == nCol) v = rv * 0.25f; }
    else if (nCol == 8) { v = rv * 0.25f; }
    bmat[e] = (_Float16)v;
  }

  float acc[G_];
  #pragma unroll
  for (int g = 0; g < G_; ++g) acc[g] = 0.f;
  float wsum = 1e-8f;
  const float invSqrtC = 0.17677669529663687f;  // 1/sqrt(32)

  for (int v = 0; v < V_-1; ++v) {
    // ---- projective warp (lane = depth) -----------------------------------
    const float* RT = ws + (size_t)(b*(V_-1) + v)*12;
    float rx = RT[0]*(float)wx + RT[1]*(float)hy + RT[2];
    float ry = RT[3]*(float)wx + RT[4]*(float)hy + RT[5];
    float rz = RT[6]*(float)wx + RT[7]*(float)hy + RT[8];
    float px = rx*dep + RT[9];
    float py = ry*dep + RT[10];
    float pz = rz*dep + RT[11];
    pz = (pz == 0.f) ? 1e-9f : pz;
    float ix = px / pz;
    float iy = py / pz;
    // (reference's normalize/denormalize round-trip is the identity)

    // ---- bilinear taps with zero padding ----------------------------------
    float x0 = floorf(ix), y0 = floorf(iy);
    float x1 = x0 + 1.f,   y1 = y0 + 1.f;
    float fx1 = ix - x0, fx0 = 1.f - fx1;
    float fy1 = iy - y0, fy0 = 1.f - fy1;
    float vx0 = (x0 >= 0.f && x0 <= (float)(W_-1)) ? 1.f : 0.f;
    float vx1 = (x1 >= 0.f && x1 <= (float)(W_-1)) ? 1.f : 0.f;
    float vy0 = (y0 >= 0.f && y0 <= (float)(H_-1)) ? 1.f : 0.f;
    float vy1 = (y1 >= 0.f && y1 <= (float)(H_-1)) ? 1.f : 0.f;
    int xc0 = (int)fminf(fmaxf(x0, 0.f), (float)(W_-1));
    int xc1 = (int)fminf(fmaxf(x1, 0.f), (float)(W_-1));
    int yc0 = (int)fminf(fmaxf(y0, 0.f), (float)(H_-1));
    int yc1 = (int)fminf(fmaxf(y1, 0.f), (float)(H_-1));
    float w00 = fx0*fy0*vx0*vy0, w01 = fx1*fy0*vx1*vy0;
    float w10 = fx0*fy1*vx0*vy1, w11 = fx1*fy1*vx1*vy1;
    int o00 = yc0*W_ + xc0, o01 = yc0*W_ + xc1;
    int o10 = yc1*W_ + xc0, o11 = yc1*W_ + xc1;

    // ---- gather + interpolate 32 channels (L2-resident) -------------------
    const float* fb = src + ((size_t)((v*B_ + b)*C_))*HW_;
    v8h abuf[4];
    _Float16* ab = (_Float16*)abuf;
    #pragma unroll
    for (int c = 0; c < C_; ++c) {
      const float* fc = fb + (size_t)c*HW_;
      float val = fc[o00]*w00 + fc[o01]*w01 + fc[o10]*w10 + fc[o11]*w11;
      ab[c] = (_Float16)val;
    }

    // ---- stage lane(=depth)-major rows to LDS, reload in WMMA A layout ----
    {
      v8h* sp = (v8h*)&Ast[wv][lane*C_];
      sp[0] = abuf[0]; sp[1] = abuf[1]; sp[2] = abuf[2]; sp[3] = abuf[3];
    }
    asm volatile("s_wait_dscnt 0" ::: "memory");

    // A 16x32, contiguous k order used here (same as B): lane L<16 row M=L holds K 0..15,
    //                        lane L>=16 row M=L-16 holds K 16..31
    const int m  = lane & 15;
    const int hh = (lane < 16) ? 0 : 16;
    const v8h* gp = (const v8h*)&Ast[wv][0];
    v8h a1lo = gp[(m*C_ + hh)        / 8];
    v8h a1hi = gp[(m*C_ + hh + 8)    / 8];
    v8h a2lo = gp[((m+16)*C_ + hh)      / 8];
    v8h a2hi = gp[((m+16)*C_ + hh + 8)  / 8];
    v16h A1, A2;
    #pragma unroll
    for (int i = 0; i < 8; ++i) {
      A1[i] = a1lo[i]; A1[8+i] = a1hi[i];
      A2[i] = a2lo[i]; A2[8+i] = a2hi[i];
    }

    // ---- the contraction: cor[g][d] and cor.sum (col 8) via WMMA ----------
    v8f Dz = {0.f,0.f,0.f,0.f,0.f,0.f,0.f,0.f};
    v8f D1 = __builtin_amdgcn_wmma_f32_16x16x32_f16(false, A1, false, bmat,
                                                    (short)0, Dz, false, false);
    v8f D2 = __builtin_amdgcn_wmma_f32_16x16x32_f16(false, A2, false, bmat,
                                                    (short)0, Dz, false, false);

    // ---- dump D (f32 16x16 layout) to LDS, reread lane=depth --------------
    const int col = lane & 15;
    const int rof = (lane < 16) ? 0 : 8;
    #pragma unroll
    for (int r = 0; r < 8; ++r) {
      Cst[wv][(r + rof)*16 + col]      = D1[r];   // depths 0..15
      Cst[wv][(r + rof + 16)*16 + col] = D2[r];   // depths 16..31
    }
    asm volatile("s_wait_dscnt 0" ::: "memory");

    const v4f* cp = (const v4f*)&Cst[wv][lane*16];
    v4f c03 = cp[0];
    v4f c47 = cp[1];
    float S = Cst[wv][lane*16 + 8];               // cor.sum over groups

    // ---- per-view attention weight: softmax over depth (wave32) -----------
    float sm  = S * 0.5f;                         // / ATTN_TEMP
    float mx  = wred_max_f32(sm);
    float ex  = __expf(sm - mx);
    float sum = wred_sum_f32(ex);
    float cw  = (ex / sum) * invSqrtC;

    wsum += cw;
    acc[0] += cw*c03.x; acc[1] += cw*c03.y; acc[2] += cw*c03.z; acc[3] += cw*c03.w;
    acc[4] += cw*c47.x; acc[5] += cw*c47.y; acc[6] += cw*c47.z; acc[7] += cw*c47.w;
  }

  // ---- epilogue: normalize, contract groups, softmax, argmax --------------
  float logit = 0.f;
  #pragma unroll
  for (int g = 0; g < G_; ++g) logit += regw[g] * (acc[g] / wsum);

  float mx   = wred_max_f32(logit);
  float ex   = __expf(logit - mx);
  float sum  = wred_sum_f32(ex);
  float attn = ex / sum;

  out[(size_t)B_*HW_ + ((size_t)(b*D_ + lane))*HW_ + pix] = attn;

  // first-max argmax over the 32 depths (cold path: shfl pair reduction)
  float bv = attn;
  int   bi = lane;
  #pragma unroll
  for (int off = 16; off > 0; off >>= 1) {
    float ov = __shfl_xor(bv, off, 32);
    int   oi = __shfl_xor(bi, off, 32);
    if (ov > bv || (ov == bv && oi < bi)) { bv = ov; bi = oi; }
  }
  float dsel = __shfl(dep, bi, 32);
  if (lane == 0) out[(size_t)b*HW_ + pix] = dsel;
}

// ---------------------------------------------------------------------------
extern "C" void kernel_launch(void* const* d_in, const int* in_sizes, int n_in,
                              void* d_out, int out_size, void* d_ws, size_t ws_size,
                              hipStream_t stream) {
  (void)in_sizes; (void)n_in; (void)out_size; (void)ws_size;
  const float* ref  = (const float*)d_in[0];
  const float* src  = (const float*)d_in[1];
  const float* proj = (const float*)d_in[2];
  const float* dh   = (const float*)d_in[3];
  const float* regw = (const float*)d_in[4];
  float* ws  = (float*)d_ws;
  float* out = (float*)d_out;

  proj_prep<<<1, 32, 0, stream>>>(proj, ws);

  const int pixels = B_ * HW_;                  // 40960 == 5120 blocks * 8 waves
  mvs_main<<<pixels / 8, 256, 0, stream>>>(ref, src, dh, regw, ws, out);
}
